// StatefulNeuronBank_67800353735023
// MI455X (gfx1250) — matrix operand order repaired and checked by reading, not compiled, not run on hardware
//
#include <hip/hip_runtime.h>
#include <hip/hip_bf16.h>

// ---------------------------------------------------------------------------
// StatefulNeuronBank fused pipeline for MI455X (gfx1250, wave32, WMMA).
//   pass0: cast x / weights f32 -> bf16
//   pass1: 4-way fused projection GEMM (v_wmma_f32_16x16x32_bf16),
//          double-buffered async global->LDS staging, nonlinearity epilogue
//   pass2: chunked parallel scan (local reduce, combine, replay)
//   pass3: output GEMM y @ W_out^T (WMMA, async staging) -> f32 out
// ---------------------------------------------------------------------------

typedef __bf16 bf16_t;
typedef __attribute__((ext_vector_type(16))) __bf16 v16bf;
typedef __attribute__((ext_vector_type(8)))  float  v8f;

#define DM     1024
#define DN     4096
#define NB     2
#define TT     4096
#define MROWS  (NB * TT)   // 8192 token rows
#define NCH    32          // scan chunks per sequence
#define CHL    (TT / NCH)  // 128 timesteps per chunk

#define AST 40   // LDS row stride (halves): 32 + 8 pad (80B, bank-conflict-free)
#define BST 40

// ---------------- workspace layout (bytes) ----------------
#define WS_XB   ((size_t)0)                                  // bf16 x       [M][DM]
#define WS_WM   (WS_XB  + (size_t)MROWS * DM * 2)
#define WS_WG   (WS_WM  + (size_t)DN * DM * 2)
#define WS_WMO  (WS_WG  + (size_t)DN * DM * 2)
#define WS_WD   (WS_WMO + (size_t)DN * DM * 2)
#define WS_WO   (WS_WD  + (size_t)DN * DM * 2)               // bf16 W_out [DM][DN]
#define WS_DEC  (WS_WO  + (size_t)DM * DN * 2)               // f32 decay  [M][DN]
#define WS_UPD  (WS_DEC + (size_t)MROWS * DN * 4)            // f32 update [M][DN]
#define WS_C1   (WS_UPD + (size_t)MROWS * DN * 4)            // bf16 c1    [M][DN]
#define WS_C2   (WS_C1  + (size_t)MROWS * DN * 2)            // bf16 c2    [M][DN]
#define WS_Y    (WS_C2  + (size_t)MROWS * DN * 2)            // bf16 y     [M][DN]
#define WS_DC   (WS_Y   + (size_t)MROWS * DN * 2)            // f32 Dc [NB][NCH][DN]
#define WS_UC   (WS_DC  + (size_t)NB * NCH * DN * 4)         // f32 Uc
#define WS_SC   (WS_UC  + (size_t)NB * NCH * DN * 4)         // f32 Sc (incoming state)

// ---------------------------------------------------------------------------
// Async global -> LDS 16B copy (CDNA5 GLOBAL_LOAD_ASYNC_TO_LDS_B128, ASYNCcnt).
// Inline asm (portable across toolchains; bypasses builtin prototype drift).
// Generic pointers to __shared__ carry the LDS byte offset in the low dword.
// ---------------------------------------------------------------------------
static __device__ __forceinline__ void async_cp16(bf16_t* dst, const bf16_t* src) {
    unsigned lds_off = (unsigned)(unsigned long long)dst;
    asm volatile("global_load_async_to_lds_b128 %0, %1, off"
                 :: "v"(lds_off), "v"(src) : "memory");
}

static __device__ __forceinline__ void wait_async0() {
    asm volatile("s_wait_asynccnt 0x0" ::: "memory");
}

// ---------------------------------------------------------------------------
__global__ __launch_bounds__(256)
void cast_f32_bf16_kernel(const float* __restrict__ src, bf16_t* __restrict__ dst, int n) {
    int i = blockIdx.x * blockDim.x + threadIdx.x;
    if (i < n) dst[i] = (bf16_t)src[i];
}

// ---------------------------------------------------------------------------
// Pass 1: four projections in one WMMA kernel + nonlinearity epilogue.
// Grid: x = DN/32 col-blocks, y = M/128 row-blocks. 256 threads = 8 waves.
// ---------------------------------------------------------------------------
__global__ __launch_bounds__(256)
void proj4_wmma_kernel(const bf16_t* __restrict__ xb,
                       const bf16_t* __restrict__ wm,  const bf16_t* __restrict__ wg,
                       const bf16_t* __restrict__ wmo, const bf16_t* __restrict__ wd,
                       const float* __restrict__ b_delta, const float* __restrict__ A_log,
                       const float* __restrict__ sw_logit,
                       float* __restrict__ dec_arr, float* __restrict__ upd_arr,
                       bf16_t* __restrict__ c1_arr, bf16_t* __restrict__ c2_arr) {
    __shared__ bf16_t As[2][128 * AST];
    __shared__ bf16_t Bs[2][4 * 32 * BST];

    const int tid  = threadIdx.x;
    const int lane = tid & 31;
    const int wave = tid >> 5;
    const int row0 = blockIdx.y * 128;
    const int col0 = blockIdx.x * 32;
    const int wrow = (wave & 3) * 32;
    const int wcol = (wave >> 2) * 16;
    const int hl   = lane & 15;
    const int half = lane >> 4;

    const bf16_t* Wp[4] = {wm, wg, wmo, wd};

    // per-thread staging assignments (2 A-chunks + 2 B-chunks, 16B each)
    const int ar0 = tid >> 2,          as0 = tid & 3;          // A chunk tid
    const int ar1 = (tid + 256) >> 2,  as1 = tid & 3;          // A chunk tid+256
    const int bp0 = tid >> 7,          bn0 = (tid & 127) >> 2, bs0 = tid & 3;
    const int bp1 = (tid + 256) >> 7,  bn1 = (tid & 127) >> 2, bs1 = tid & 3;

    auto stage = [&](int buf, int kb) {
        async_cp16(&As[buf][ar0 * AST + as0 * 8], xb + (size_t)(row0 + ar0) * DM + kb + as0 * 8);
        async_cp16(&As[buf][ar1 * AST + as1 * 8], xb + (size_t)(row0 + ar1) * DM + kb + as1 * 8);
        async_cp16(&Bs[buf][bp0 * 32 * BST + bn0 * BST + bs0 * 8],
                   Wp[bp0] + (size_t)(col0 + bn0) * DM + kb + bs0 * 8);
        async_cp16(&Bs[buf][bp1 * 32 * BST + bn1 * BST + bs1 * 8],
                   Wp[bp1] + (size_t)(col0 + bn1) * DM + kb + bs1 * 8);
    };

    const v8f vzero = {0.f, 0.f, 0.f, 0.f, 0.f, 0.f, 0.f, 0.f};
    v8f acc[4][2];
    for (int p = 0; p < 4; ++p)
        for (int rt = 0; rt < 2; ++rt) acc[p][rt] = vzero;

    stage(0, 0);
    int cur = 0;
    for (int kb = 0; kb < DM; kb += 32) {
        wait_async0();          // our async copies into buf `cur` have landed
        __syncthreads();        // everyone's copies have landed
        if (kb + 32 < DM) stage(cur ^ 1, kb + 32);   // overlap next copy with math

        union AF { v16bf v; uint4 u[2]; };
        AF af[2];
        const int koff = half * 8;
        for (int rt = 0; rt < 2; ++rt) {
            const bf16_t* ap = &As[cur][(wrow + rt * 16 + hl) * AST];
            af[rt].u[0] = *(const uint4*)(ap + koff);
            af[rt].u[1] = *(const uint4*)(ap + 16 + koff);
        }
        const int kb2 = half * 8;
        AF bfr[4];
        for (int p = 0; p < 4; ++p) {
            const bf16_t* bp = &Bs[cur][p * 32 * BST + (wcol + hl) * BST + kb2];
            bfr[p].u[0] = *(const uint4*)(bp);
            bfr[p].u[1] = *(const uint4*)(bp + 16);
        }
        for (int p = 0; p < 4; ++p)
            for (int rt = 0; rt < 2; ++rt)
                acc[p][rt] = __builtin_amdgcn_wmma_f32_16x16x32_bf16(
                    false, af[rt].v, false, bfr[p].v, (short)0, acc[p][rt], false, false);
        __syncthreads();        // all reads of buf `cur` done before it is re-staged
        cur ^= 1;
    }

    // epilogue: C layout VGPR r -> M=r (+8 for lanes>=16), N=lane&15
    const int col = col0 + wcol + hl;
    const float bd  = b_delta[col];
    const float nA  = -__expf(A_log[col]);
    const float sw  = 1.f / (1.f + __expf(-sw_logit[col]));
    for (int rt = 0; rt < 2; ++rt) {
        for (int r = 0; r < 8; ++r) {
            const float raw  = acc[0][rt][r];
            const float gate = 1.f / (1.f + __expf(-acc[1][rt][r]));
            const float mod  = tanhf(acc[2][rt][r]);
            const float zd   = acc[3][rt][r] + bd;
            const float delta = (zd > 20.f) ? zd : log1pf(__expf(zd));
            const float dec  = __expf(delta * nA);
            const float upd  = (1.f - dec) * gate * tanhf(raw);
            const float gelu = 0.5f * raw * (1.f + erff(raw * 0.70710678f));
            const float opm  = 1.f + mod;
            const int   row  = row0 + wrow + rt * 16 + r + half * 8;
            const size_t idx = (size_t)row * DN + col;
            dec_arr[idx] = dec;
            upd_arr[idx] = upd;
            c1_arr[idx]  = (bf16_t)((1.f - sw) * gelu * opm);
            c2_arr[idx]  = (bf16_t)(sw * opm);
        }
    }
}

// ---------------------------------------------------------------------------
// Pass 2a: per-chunk local scan reduce: D = prod(d), U = scan-from-zero.
// One thread per (b, chunk, n): 262144 threads -> 8192 waves of latency hiding.
// ---------------------------------------------------------------------------
__global__ __launch_bounds__(256)
void scan_local_kernel(const float* __restrict__ dec_arr, const float* __restrict__ upd_arr,
                       float* __restrict__ Dc, float* __restrict__ Uc) {
    const int gid = blockIdx.x * blockDim.x + threadIdx.x;     // [(b*NCH+ch)*DN + n]
    if (gid >= NB * NCH * DN) return;
    const int n    = gid & (DN - 1);
    const int rest = gid >> 12;            // DN = 4096
    const int ch   = rest & (NCH - 1);
    const int b    = rest >> 5;            // NCH = 32
    size_t idx = ((size_t)b * TT + (size_t)ch * CHL) * DN + n;
    float D = 1.f, U = 0.f;
    for (int t = 0; t < CHL; ++t, idx += DN) {
        const float d = dec_arr[idx];
        const float u = upd_arr[idx];
        D *= d;
        U = d * U + u;
    }
    Dc[gid] = D;
    Uc[gid] = U;
}

// ---------------------------------------------------------------------------
// Pass 2b: sequential combine over chunks; emits incoming state per chunk
// and the final state output.  8192 threads, 32 steps each.
// ---------------------------------------------------------------------------
__global__ __launch_bounds__(256)
void scan_combine_kernel(const float* __restrict__ Dc, const float* __restrict__ Uc,
                         const float* __restrict__ state_in,
                         float* __restrict__ Sc, float* __restrict__ state_out) {
    const int gid = blockIdx.x * blockDim.x + threadIdx.x;     // [0, NB*DN)
    if (gid >= NB * DN) return;
    const int n = gid & (DN - 1);
    const int b = gid >> 12;
    float S = state_in[gid];
    for (int ch = 0; ch < NCH; ++ch) {
        const size_t j = ((size_t)b * NCH + ch) * DN + n;
        Sc[j] = S;
        S = Dc[j] * S + Uc[j];
    }
    state_out[gid] = S;
}

// ---------------------------------------------------------------------------
// Pass 2c: replay each chunk from its incoming state, producing y (bf16).
// ---------------------------------------------------------------------------
__global__ __launch_bounds__(256)
void scan_apply_kernel(const float* __restrict__ dec_arr, const float* __restrict__ upd_arr,
                       const bf16_t* __restrict__ c1_arr, const bf16_t* __restrict__ c2_arr,
                       const float* __restrict__ Sc, bf16_t* __restrict__ y_arr) {
    const int gid = blockIdx.x * blockDim.x + threadIdx.x;
    if (gid >= NB * NCH * DN) return;
    const int n    = gid & (DN - 1);
    const int rest = gid >> 12;
    const int ch   = rest & (NCH - 1);
    const int b    = rest >> 5;
    size_t idx = ((size_t)b * TT + (size_t)ch * CHL) * DN + n;
    float s = Sc[gid];
    for (int t = 0; t < CHL; ++t, idx += DN) {
        s = dec_arr[idx] * s + upd_arr[idx];
        y_arr[idx] = (bf16_t)((float)c1_arr[idx] + (float)c2_arr[idx] * s);
    }
}

// ---------------------------------------------------------------------------
// Pass 3: out = y @ W_out^T.  W_out stored [DM][DN] row-major == [N][K].
// ---------------------------------------------------------------------------
__global__ __launch_bounds__(256)
void outproj_wmma_kernel(const bf16_t* __restrict__ yb, const bf16_t* __restrict__ wo,
                         float* __restrict__ out) {
    __shared__ bf16_t As[2][128 * AST];
    __shared__ bf16_t Bs[2][32 * BST];

    const int tid  = threadIdx.x;
    const int lane = tid & 31;
    const int wave = tid >> 5;
    const int row0 = blockIdx.y * 128;
    const int col0 = blockIdx.x * 32;
    const int wrow = (wave & 3) * 32;
    const int wcol = (wave >> 2) * 16;
    const int hl   = lane & 15;
    const int half = lane >> 4;

    const int ar0 = tid >> 2,         as0 = tid & 3;
    const int ar1 = (tid + 256) >> 2, as1 = tid & 3;
    const int bn0 = tid >> 2,         bs0 = tid & 3;          // valid for tid < 128

    auto stage = [&](int buf, int kb) {
        async_cp16(&As[buf][ar0 * AST + as0 * 8], yb + (size_t)(row0 + ar0) * DN + kb + as0 * 8);
        async_cp16(&As[buf][ar1 * AST + as1 * 8], yb + (size_t)(row0 + ar1) * DN + kb + as1 * 8);
        if (tid < 128)
            async_cp16(&Bs[buf][bn0 * BST + bs0 * 8], wo + (size_t)(col0 + bn0) * DN + kb + bs0 * 8);
    };

    const v8f vzero = {0.f, 0.f, 0.f, 0.f, 0.f, 0.f, 0.f, 0.f};
    v8f acc[2] = {vzero, vzero};

    stage(0, 0);
    int cur = 0;
    for (int kb = 0; kb < DN; kb += 32) {
        wait_async0();
        __syncthreads();
        if (kb + 32 < DN) stage(cur ^ 1, kb + 32);

        union AF { v16bf v; uint4 u[2]; };
        AF af[2], bfr;
        const int koff = half * 8;
        for (int rt = 0; rt < 2; ++rt) {
            const bf16_t* ap = &As[cur][(wrow + rt * 16 + hl) * AST];
            af[rt].u[0] = *(const uint4*)(ap + koff);
            af[rt].u[1] = *(const uint4*)(ap + 16 + koff);
        }
        const bf16_t* bp = &Bs[cur][(wcol + hl) * BST + half * 8];
        bfr.u[0] = *(const uint4*)(bp);
        bfr.u[1] = *(const uint4*)(bp + 16);

        for (int rt = 0; rt < 2; ++rt)
            acc[rt] = __builtin_amdgcn_wmma_f32_16x16x32_bf16(
                false, af[rt].v, false, bfr.v, (short)0, acc[rt], false, false);
        __syncthreads();
        cur ^= 1;
    }

    const int col = col0 + wcol + hl;
    for (int rt = 0; rt < 2; ++rt)
        for (int r = 0; r < 8; ++r) {
            const int row = row0 + wrow + rt * 16 + r + half * 8;
            out[(size_t)row * DM + col] = acc[rt][r];
        }
}

// ---------------------------------------------------------------------------
extern "C" void kernel_launch(void* const* d_in, const int* in_sizes, int n_in,
                              void* d_out, int out_size, void* d_ws, size_t ws_size,
                              hipStream_t stream) {
    (void)in_sizes; (void)n_in; (void)out_size; (void)ws_size;
    const float* x        = (const float*)d_in[0];
    const float* state    = (const float*)d_in[1];
    const float* W_main   = (const float*)d_in[2];
    const float* W_gate   = (const float*)d_in[3];
    const float* W_mod    = (const float*)d_in[4];
    const float* W_out    = (const float*)d_in[5];
    const float* W_delta  = (const float*)d_in[6];
    const float* b_delta  = (const float*)d_in[7];
    const float* A_log    = (const float*)d_in[8];
    const float* sw_logit = (const float*)d_in[9];

    char* ws = (char*)d_ws;
    bf16_t* xb   = (bf16_t*)(ws + WS_XB);
    bf16_t* wm   = (bf16_t*)(ws + WS_WM);
    bf16_t* wg   = (bf16_t*)(ws + WS_WG);
    bf16_t* wmo  = (bf16_t*)(ws + WS_WMO);
    bf16_t* wd   = (bf16_t*)(ws + WS_WD);
    bf16_t* wo   = (bf16_t*)(ws + WS_WO);
    float*  decA = (float*)(ws + WS_DEC);
    float*  updA = (float*)(ws + WS_UPD);
    bf16_t* c1A  = (bf16_t*)(ws + WS_C1);
    bf16_t* c2A  = (bf16_t*)(ws + WS_C2);
    bf16_t* yA   = (bf16_t*)(ws + WS_Y);
    float*  DcA  = (float*)(ws + WS_DC);
    float*  UcA  = (float*)(ws + WS_UC);
    float*  ScA  = (float*)(ws + WS_SC);

    float* out_f   = (float*)d_out;                  // [M][DM]
    float* state_f = out_f + (size_t)MROWS * DM;     // [NB][DN] final state

    const int nx = MROWS * DM, nw = DN * DM;
    cast_f32_bf16_kernel<<<(nx + 255) / 256, 256, 0, stream>>>(x, xb, nx);
    cast_f32_bf16_kernel<<<(nw + 255) / 256, 256, 0, stream>>>(W_main,  wm,  nw);
    cast_f32_bf16_kernel<<<(nw + 255) / 256, 256, 0, stream>>>(W_gate,  wg,  nw);
    cast_f32_bf16_kernel<<<(nw + 255) / 256, 256, 0, stream>>>(W_mod,   wmo, nw);
    cast_f32_bf16_kernel<<<(nw + 255) / 256, 256, 0, stream>>>(W_delta, wd,  nw);
    cast_f32_bf16_kernel<<<(nw + 255) / 256, 256, 0, stream>>>(W_out,   wo,  nw);

    proj4_wmma_kernel<<<dim3(DN / 32, MROWS / 128), 256, 0, stream>>>(
        xb, wm, wg, wmo, wd, b_delta, A_log, sw_logit, decA, updA, c1A, c2A);

    scan_local_kernel<<<(NB * NCH * DN) / 256, 256, 0, stream>>>(decA, updA, DcA, UcA);
    scan_combine_kernel<<<(NB * DN) / 256, 256, 0, stream>>>(DcA, UcA, state, ScA, state_f);
    scan_apply_kernel<<<(NB * NCH * DN) / 256, 256, 0, stream>>>(decA, updA, c1A, c2A, ScA, yA);

    outproj_wmma_kernel<<<dim3(DM / 32, MROWS / 128), 256, 0, stream>>>(yA, wo, out_f);
}
